// DTI_Sparse_MoE_63376537420133
// MI455X (gfx1250) — matrix operand order repaired and checked by reading, not compiled, not run on hardware
//
#include <hip/hip_runtime.h>
#include <math.h>

// ---------------------------------------------------------------------------
// DTI Sparse-MoE forward for gfx1250 (MI455X). All GEMM-shaped math uses
// V_WMMA_F32_16X16X32_F16 (wave32 fragment layouts per CDNA5 ISA 7.12.2).
// GEMMs use a 64x64 block tile: the A slab is staged in LDS via
// GLOBAL_LOAD_ASYNC_TO_LDS_B128 (ASYNCcnt-tracked), each wave owns a
// 16-wide N strip and accumulates four 16x16 WMMA tiles, reusing one B
// fragment across 4 WMMAs. Attention stages scores / probabilities in LDS
// between its two WMMA stages.
// ---------------------------------------------------------------------------

#define D_MODEL   128
#define N_HEAD    4
#define HEAD_DIM  32
#define FF_DIM    256
#define CHUNK     512
#define NEXP      6
#define BATCH     64
#define L_DRUG    128
#define L_PROT    1024
#define FEAT      512
#define EXP_HID   1024

typedef __attribute__((ext_vector_type(16))) _Float16 v16h;
typedef __attribute__((ext_vector_type(8)))  float    v8f;
typedef __attribute__((ext_vector_type(4)))  float    v4f;
typedef int vi4 __attribute__((vector_size(16)));   // matches builtin proto

#if defined(__has_builtin)
#if __has_builtin(__builtin_amdgcn_global_load_async_to_lds_b128) && \
    __has_builtin(__builtin_amdgcn_s_wait_asynccnt)
#define USE_ASYNC_LDS 1
#endif
#endif
#ifndef USE_ASYNC_LDS
#define USE_ASYNC_LDS 0
#endif

#define AS1 __attribute__((address_space(1)))
#define AS3 __attribute__((address_space(3)))

// ---------------- WMMA fragment loaders (wave32 layouts, ISA 7.12.2) -------
// A (16x32 f16): lane l<16 -> row M=l, halves 0..7 = K kb..kb+7, 8..15 = 16+kb..
__device__ __forceinline__ v16h ld_a_f32(const float* __restrict__ A, int lda,
                                         int lane, int k0) {
  const float* p = A + (size_t)(lane & 15) * lda + k0 + ((lane >> 4) << 3);
  v16h a;
#pragma unroll
  for (int i = 0; i < 8; ++i) {
    a[i]     = (_Float16)p[i];
    a[8 + i] = (_Float16)p[16 + i];
  }
  return a;
}

// A fragment from a 16x32 f32 tile in LDS (ld = 32)
__device__ __forceinline__ v16h ld_a_lds(const float* At, int lane) {
  const float* p = At + (lane & 15) * 32 + ((lane >> 4) << 3);
  v16h a;
#pragma unroll
  for (int i = 0; i < 8; ++i) {
    a[i]     = (_Float16)p[i];
    a[8 + i] = (_Float16)p[16 + i];
  }
  return a;
}

// B (32x16 f16), source stored [N,K] (PyTorch weight): frag(k,n) = B[n][k]
__device__ __forceinline__ v16h ld_bt_f32(const float* __restrict__ B, int ldb,
                                          int lane, int k0) {
  const float* p = B + (size_t)(lane & 15) * ldb + k0 + ((lane >> 4) << 3);
  v16h b;
#pragma unroll
  for (int i = 0; i < 8; ++i) {
    b[i]     = (_Float16)p[i];
    b[8 + i] = (_Float16)p[16 + i];
  }
  return b;
}

// B (32x16 f16), source stored [K,N]: frag(k,n) = B[k][n]
__device__ __forceinline__ v16h ld_bn_f32(const float* __restrict__ B, int ldb,
                                          int lane, int k0) {
  const float* p = B + (size_t)(k0 + ((lane >> 4) << 3)) * ldb + (lane & 15);
  v16h b;
#pragma unroll
  for (int i = 0; i < 8; ++i) {
    b[i]     = (_Float16)p[(size_t)i * ldb];
    b[8 + i] = (_Float16)p[(size_t)(16 + i) * ldb];
  }
  return b;
}

// Stage a 64x32 f32 A slab into LDS (128 threads, 16 floats each)
__device__ __forceinline__ void stage_a_slab(const float* __restrict__ Ab,
                                             float* at, int K, int k0) {
  int row = threadIdx.x >> 1;
  int col = (threadIdx.x & 1) << 4;
  const float* src = Ab + (size_t)row * K + k0 + col;
  float* dst = at + row * 32 + col;
#if USE_ASYNC_LDS
#pragma unroll
  for (int j = 0; j < 4; ++j)
    __builtin_amdgcn_global_load_async_to_lds_b128(
        (AS1 vi4*)(src + 4 * j), (AS3 vi4*)(dst + 4 * j), 0, 0);
  __builtin_amdgcn_s_wait_asynccnt(0);
#else
#pragma unroll
  for (int j = 0; j < 4; ++j) *(v4f*)(dst + 4 * j) = *(const v4f*)(src + 4 * j);
#endif
}

// ---------------- GEMM: C[M,N] = act(A[M,K] @ B[N,K]^T + bias) -------------
// grid = (M/64, N/64), 128 threads. Wave w owns N strip n0+16w and the four
// 16-row M subtiles: 1 B fragment feeds 4 WMMAs per K step.
__global__ __launch_bounds__(128) void gemm_nt_kernel(
    const float* __restrict__ A, const float* __restrict__ B,
    const float* __restrict__ bias, float* __restrict__ C,
    int M, int N, int K, int act) {
  __shared__ float at[64 * 32];
  int lane = threadIdx.x & 31;
  int wave = threadIdx.x >> 5;
  int m0 = blockIdx.x << 6;
  int n0 = (blockIdx.y << 6) + (wave << 4);
  const float* Ab = A + (size_t)m0 * K;
  const float* Bb = B + (size_t)n0 * K;
  v8f zero = {0.f, 0.f, 0.f, 0.f, 0.f, 0.f, 0.f, 0.f};
  v8f acc[4];
#pragma unroll
  for (int s = 0; s < 4; ++s) acc[s] = zero;
  for (int k0 = 0; k0 < K; k0 += 32) {
    stage_a_slab(Ab, at, K, k0);
    if (k0 + 32 < K)                        // hint next weight slab
      __builtin_prefetch(Bb + (size_t)(lane & 15) * K + k0 + 32, 0, 1);
    __syncthreads();
    v16h b = ld_bt_f32(Bb, K, lane, k0);
#pragma unroll
    for (int s = 0; s < 4; ++s) {
      v16h a = ld_a_lds(&at[s * 16 * 32], lane);
      acc[s] = __builtin_amdgcn_wmma_f32_16x16x32_f16(false, a, false, b,
                                                      (short)0, acc[s],
                                                      false, false);
    }
    __syncthreads();
  }
  int n  = n0 + (lane & 15);
  int mb = (lane >> 4) << 3;
  float bv = bias ? bias[n] : 0.f;
#pragma unroll
  for (int s = 0; s < 4; ++s)
#pragma unroll
    for (int r = 0; r < 8; ++r) {
      float v = acc[s][r] + bv;
      if (act == 1) v = fmaxf(v, 0.f);
      C[(size_t)(m0 + s * 16 + mb + r) * N + n] = v;
    }
}

// ---------------- GEMM: C[M,N] = act(A[M,K] @ B[K,N] + bias) ---------------
__global__ __launch_bounds__(128) void gemm_nn_kernel(
    const float* __restrict__ A, const float* __restrict__ B,
    const float* __restrict__ bias, float* __restrict__ C,
    int M, int N, int K, int act) {
  __shared__ float at[64 * 32];
  int lane = threadIdx.x & 31;
  int wave = threadIdx.x >> 5;
  int m0 = blockIdx.x << 6;
  int n0 = (blockIdx.y << 6) + (wave << 4);
  const float* Ab = A + (size_t)m0 * K;
  const float* Bb = B + n0;
  v8f zero = {0.f, 0.f, 0.f, 0.f, 0.f, 0.f, 0.f, 0.f};
  v8f acc[4];
#pragma unroll
  for (int s = 0; s < 4; ++s) acc[s] = zero;
  for (int k0 = 0; k0 < K; k0 += 32) {
    stage_a_slab(Ab, at, K, k0);
    __syncthreads();
    v16h b = ld_bn_f32(Bb, N, lane, k0);
#pragma unroll
    for (int s = 0; s < 4; ++s) {
      v16h a = ld_a_lds(&at[s * 16 * 32], lane);
      acc[s] = __builtin_amdgcn_wmma_f32_16x16x32_f16(false, a, false, b,
                                                      (short)0, acc[s],
                                                      false, false);
    }
    __syncthreads();
  }
  int n  = n0 + (lane & 15);
  int mb = (lane >> 4) << 3;
  float bv = bias ? bias[n] : 0.f;
#pragma unroll
  for (int s = 0; s < 4; ++s)
#pragma unroll
    for (int r = 0; r < 8; ++r) {
      float v = acc[s][r] + bv;
      if (act == 1) v = fmaxf(v, 0.f);
      C[(size_t)(m0 + s * 16 + mb + r) * N + n] = v;
    }
}

// ---------------- Fused attention (one 16-query tile / head / sequence) ----
// qkv: [NC*S, 384] (q|k|v per head). O: [NC*S, 128]. LDS: 16*S f32 + 16*S f16
__global__ __launch_bounds__(128) void attention_kernel(
    const float* __restrict__ qkv, const int* __restrict__ kmask,
    float* __restrict__ O, int S) {
  extern __shared__ float smemf[];
  float*    sc = smemf;                                  // [16][S] scores
  _Float16* ah = (_Float16*)(smemf + 16 * S);            // [16][S] probs f16
  int lane = threadIdx.x & 31;
  int wave = threadIdx.x >> 5;
  int qt  = blockIdx.x;
  int h   = blockIdx.y;
  int seq = blockIdx.z;
  const float* base = qkv + (size_t)seq * S * 384;
  const float* Qb = base + (size_t)(qt * 16) * 384 + h * HEAD_DIM;
  const float* Kb = base + D_MODEL     + h * HEAD_DIM;
  const float* Vb = base + 2 * D_MODEL + h * HEAD_DIM;
  const int* mrow = kmask + (size_t)seq * S;
  const int nt = S >> 4;
  const float scale = 0.17677669529663687f;              // 1/sqrt(32)

  // Phase 1: scores = Q K^T * scale  (K-dim = 32 -> single WMMA per tile)
  v16h aq = ld_a_f32(Qb, 384, lane, 0);
  for (int jt = wave; jt < nt; jt += 4) {
    v16h b = ld_bt_f32(Kb + (size_t)(jt * 16) * 384, 384, lane, 0);
    v8f c = {0.f, 0.f, 0.f, 0.f, 0.f, 0.f, 0.f, 0.f};
    c = __builtin_amdgcn_wmma_f32_16x16x32_f16(false, aq, false, b,
                                               (short)0, c, false, false);
    int j  = jt * 16 + (lane & 15);
    int mb = (lane >> 4) << 3;
    float mv = mrow[j] ? -1e9f : 0.f;
#pragma unroll
    for (int r = 0; r < 8; ++r) sc[(mb + r) * S + j] = c[r] * scale + mv;
  }
  __syncthreads();

  // Phase 2: row softmax, repack probabilities as f16 in LDS
  if (threadIdx.x < 16) {
    int r = threadIdx.x;
    float mx = -3.4e38f;
    for (int j = 0; j < S; ++j) mx = fmaxf(mx, sc[r * S + j]);
    float sum = 0.f;
    for (int j = 0; j < S; ++j) {
      float e = expf(sc[r * S + j] - mx);
      sc[r * S + j] = e;
      sum += e;
    }
    float inv = 1.f / sum;
    for (int j = 0; j < S; ++j) ah[r * S + j] = (_Float16)(sc[r * S + j] * inv);
  }
  __syncthreads();

  // Phase 3: O = P V  (waves 0/1 own the two 16-wide halves of HEAD_DIM)
  if (wave < 2) {
    v8f c = {0.f, 0.f, 0.f, 0.f, 0.f, 0.f, 0.f, 0.f};
    for (int kk = 0; kk < S; kk += 32) {
      v16h a;
      int m = lane & 15, kb = (lane >> 4) << 3;
#pragma unroll
      for (int i = 0; i < 8; ++i) {
        a[i]     = ah[m * S + kk + kb + i];
        a[8 + i] = ah[m * S + kk + 16 + kb + i];
      }
      v16h b = ld_bn_f32(Vb + (size_t)kk * 384 + wave * 16, 384, lane, 0);
      c = __builtin_amdgcn_wmma_f32_16x16x32_f16(false, a, false, b,
                                                 (short)0, c, false, false);
    }
    int d  = wave * 16 + (lane & 15);
    int mb = (lane >> 4) << 3;
#pragma unroll
    for (int r = 0; r < 8; ++r)
      O[(size_t)(seq * S + qt * 16 + mb + r) * D_MODEL + h * HEAD_DIM + d] = c[r];
  }
}

// ---------------- Embedding + sinusoidal PE --------------------------------
__global__ void embed_kernel(const int* __restrict__ tok,
                             const float* __restrict__ emb,
                             float* __restrict__ x, int L) {
  int t = blockIdx.x;
  int d = threadIdx.x;
  int pos = t % L;
  int i2 = d & ~1;
  float div = expf((float)i2 * (-9.210340371976184f / 128.f)); // -ln(10000)/128
  float pe = (d & 1) ? cosf(pos * div) : sinf(pos * div);
  x[(size_t)t * D_MODEL + d] = emb[(size_t)tok[t] * D_MODEL + d] + pe;
}

// ---------------- fix_all_pad ----------------------------------------------
__global__ void fix_mask_kernel(const int* __restrict__ m, int* __restrict__ out,
                                int rows, int S) {
  int r = blockIdx.x * blockDim.x + threadIdx.x;
  if (r >= rows) return;
  int all = 1;
  for (int j = 0; j < S; ++j) {
    int v = m[(size_t)r * S + j];
    out[(size_t)r * S + j] = v;
    all &= (v != 0);
  }
  if (all) out[(size_t)r * S] = 0;
}

// ---------------- LayerNorm (dim 128), optional residual, wave per row -----
__global__ __launch_bounds__(128) void ln_kernel(
    const float* __restrict__ x, const float* __restrict__ res,
    const float* __restrict__ g, const float* __restrict__ b,
    float* __restrict__ out, int T) {
  int lane = threadIdx.x & 31;
  int row = blockIdx.x * 4 + (threadIdx.x >> 5);
  if (row >= T) return;                      // wave-uniform
  const float* xr = x + (size_t)row * D_MODEL;
  const float* rr = res ? res + (size_t)row * D_MODEL : nullptr;
  float v[4];
#pragma unroll
  for (int i = 0; i < 4; ++i) {
    int d = lane + 32 * i;
    v[i] = xr[d] + (rr ? rr[d] : 0.f);
  }
  float s = v[0] + v[1] + v[2] + v[3];
#pragma unroll
  for (int o = 16; o > 0; o >>= 1) s += __shfl_xor(s, o, 32);
  float mean = s * (1.f / 128.f);
  float q = 0.f;
#pragma unroll
  for (int i = 0; i < 4; ++i) { float dd = v[i] - mean; q += dd * dd; }
#pragma unroll
  for (int o = 16; o > 0; o >>= 1) q += __shfl_xor(q, o, 32);
  float inv = rsqrtf(q * (1.f / 128.f) + 1e-5f);
#pragma unroll
  for (int i = 0; i < 4; ++i) {
    int d = lane + 32 * i;
    out[(size_t)row * D_MODEL + d] = (v[i] - mean) * inv * g[d] + b[d];
  }
}

// ---------------- zero padded tokens (original mask) -----------------------
__global__ void mask_zero_kernel(float* __restrict__ x,
                                 const int* __restrict__ m, int T) {
  int t = blockIdx.x;
  if (m[t]) x[(size_t)t * D_MODEL + threadIdx.x] = 0.f;
}

// ---------------- attention-pool score: s = tanh(h) . w2 + b2 --------------
__global__ void pool_score_kernel(const float* __restrict__ h,
                                  const float* __restrict__ w2,
                                  const float* __restrict__ b2,
                                  float* __restrict__ s, int T) {
  int t = blockIdx.x * blockDim.x + threadIdx.x;
  if (t >= T) return;
  float acc = b2[0];
  for (int j = 0; j < 64; ++j) acc += tanhf(h[(size_t)t * 64 + j]) * w2[j];
  s[t] = acc;
}

// ---------------- masked softmax pool over sequence ------------------------
__global__ __launch_bounds__(128) void pool_kernel(
    const float* __restrict__ x, const float* __restrict__ s,
    const int* __restrict__ mask, float* __restrict__ rep,
    int L, int colOff) {
  __shared__ float w[1024];
  int bb = blockIdx.x;
  if (threadIdx.x == 0) {
    float mx = -3.4e38f; int any = 0;
    for (int l = 0; l < L; ++l)
      if (!mask[(size_t)bb * L + l]) { any = 1; mx = fmaxf(mx, s[(size_t)bb * L + l]); }
    float sum = 0.f;
    for (int l = 0; l < L; ++l) {
      float e = (any && !mask[(size_t)bb * L + l]) ? expf(s[(size_t)bb * L + l] - mx) : 0.f;
      w[l] = e; sum += e;
    }
    float inv = sum > 0.f ? 1.f / sum : 0.f;      // nan_to_num path
    for (int l = 0; l < L; ++l) w[l] *= inv;
  }
  __syncthreads();
  int d = threadIdx.x;
  float acc = 0.f;
  for (int l = 0; l < L; ++l) acc += w[l] * x[((size_t)bb * L + l) * D_MODEL + d];
  rep[(size_t)bb * 256 + colOff + d] = acc;
}

// ---------------- gate: relu(h)@w2^T+b2 -> softmax -> top-2 weights --------
__global__ void gate_kernel(const float* __restrict__ hn,
                            const float* __restrict__ w2,
                            const float* __restrict__ b2,
                            float* __restrict__ probs,
                            float* __restrict__ wmat) {
  int bidx = threadIdx.x;
  if (bidx >= BATCH) return;
  float lg[NEXP];
  for (int e = 0; e < NEXP; ++e) {
    float acc = b2[e];
    for (int d = 0; d < D_MODEL; ++d)
      acc += fmaxf(hn[(size_t)bidx * D_MODEL + d], 0.f) * w2[(size_t)e * D_MODEL + d];
    lg[e] = acc;
  }
  float mx = lg[0];
  for (int e = 1; e < NEXP; ++e) mx = fmaxf(mx, lg[e]);
  float sum = 0.f;
  for (int e = 0; e < NEXP; ++e) { lg[e] = expf(lg[e] - mx); sum += lg[e]; }
  for (int e = 0; e < NEXP; ++e) { lg[e] /= sum; probs[bidx * NEXP + e] = lg[e]; }
  int i0 = 0;
  for (int e = 1; e < NEXP; ++e) if (lg[e] > lg[i0]) i0 = e;
  int i1 = -1;
  for (int e = 0; e < NEXP; ++e) {
    if (e == i0) continue;
    if (i1 < 0 || lg[e] > lg[i1]) i1 = e;
  }
  float tv = lg[i0] + lg[i1];
  for (int e = 0; e < NEXP; ++e) wmat[bidx * NEXP + e] = 0.f;
  wmat[bidx * NEXP + i0] = lg[i0] / tv;
  wmat[bidx * NEXP + i1] = lg[i1] / tv;
}

// ---------------- expert second layer: eo[e,b] = eh[e,b,:].w2[e] + b2[e] ---
__global__ __launch_bounds__(32) void expert_reduce_kernel(
    const float* __restrict__ eh, const float* __restrict__ w2,
    const float* __restrict__ b2, float* __restrict__ eo) {
  int bidx = blockIdx.x, e = blockIdx.y, lane = threadIdx.x;
  const float* row = eh + ((size_t)e * BATCH + bidx) * EXP_HID;
  const float* wv  = w2 + (size_t)e * EXP_HID;
  float acc = 0.f;
  for (int hh = lane; hh < EXP_HID; hh += 32) acc += row[hh] * wv[hh];
  for (int o = 16; o > 0; o >>= 1) acc += __shfl_xor(acc, o, 32);
  if (lane == 0) eo[e * BATCH + bidx] = acc + b2[e];
}

// ---------------- aggregation MLP ------------------------------------------
__global__ void final_kernel(const float* __restrict__ eo,
                             const float* __restrict__ wmat,
                             const float* __restrict__ aw1,
                             const float* __restrict__ ab1,
                             const float* __restrict__ aw2,
                             const float* __restrict__ ab2,
                             float* __restrict__ out) {
  int b = threadIdx.x;
  if (b >= BATCH) return;
  float em[NEXP];
  for (int e = 0; e < NEXP; ++e) em[e] = eo[e * BATCH + b] * wmat[b * NEXP + e];
  float acc = ab2[0];
  for (int j = 0; j < 64; ++j) {
    float hv = ab1[j];
    for (int e = 0; e < NEXP; ++e) hv += em[e] * aw1[j * NEXP + e];
    acc += fmaxf(hv, 0.f) * aw2[j];
  }
  out[b] = acc;
}

// ---------------- CV^2 load-balancing aux loss -----------------------------
__global__ void aux_kernel(const float* __restrict__ probs, float* __restrict__ out) {
  if (threadIdx.x != 0) return;
  float imp[NEXP];
  for (int e = 0; e < NEXP; ++e) {
    float s = 0.f;
    for (int b = 0; b < BATCH; ++b) s += probs[b * NEXP + e];
    imp[e] = s;
  }
  float mean = 0.f;
  for (int e = 0; e < NEXP; ++e) mean += imp[e];
  mean /= (float)NEXP;
  float var = 0.f;
  for (int e = 0; e < NEXP; ++e) { float d = imp[e] - mean; var += d * d; }
  var /= (float)NEXP;
  out[BATCH] = 0.1f * (var / (mean * mean + 1e-10f));
}

// ===========================================================================
// Host side
// ===========================================================================

// Input flattening (setup_inputs dict, recursive insertion order):
// 0 drug_emb, 1 prot_emb,
// 2..13  drug_layers[0]: wqkv,bqkv,wo,bo,ln1_g,ln1_b,w1,b1,w2,b2,ln2_g,ln2_b
// 14..25 drug_layers[1], 26..37 prot_layers[0], 38..49 prot_layers[1],
// 50..53 drug_pool(w1,b1,w2,b2), 54..57 prot_pool,
// 58..63 gate(w1,b1,ln_g,ln_b,w2,b2), 64..67 experts(w1,b1,w2,b2),
// 68..71 agg(w1,b1,w2,b2), 72 expert_feats, 73 drug_tokens, 74 prot_tokens,
// 75 drug_mask, 76 prot_mask
enum {
  IX_DEMB = 0, IX_PEMB = 1, IX_DL0 = 2, IX_DL1 = 14, IX_PL0 = 26, IX_PL1 = 38,
  IX_DPOOL = 50, IX_PPOOL = 54, IX_GATE = 58, IX_EXP = 64, IX_AGG = 68,
  IX_FEATS = 72, IX_DTOK = 73, IX_PTOK = 74, IX_DMASK = 75, IX_PMASK = 76
};

static inline void launch_gemm_nt(hipStream_t st, const float* A, const float* B,
                                  const float* bias, float* C,
                                  int M, int N, int K, int act) {
  dim3 g(M / 64, N / 64);
  gemm_nt_kernel<<<g, 128, 0, st>>>(A, B, bias, C, M, N, K, act);
}

static inline void launch_gemm_nn(hipStream_t st, const float* A, const float* B,
                                  const float* bias, float* C,
                                  int M, int N, int K, int act) {
  dim3 g(M / 64, N / 64);
  gemm_nn_kernel<<<g, 128, 0, st>>>(A, B, bias, C, M, N, K, act);
}

static void run_encoder(hipStream_t st, float* x, int T, int S, int NC,
                        const int* mfix, const int* morig,
                        float* qkv, float* ba, float* bb,
                        void* const* d_in, const int layerBase[2]) {
  for (int li = 0; li < 2; ++li) {
    int base = layerBase[li];
    const float* wqkv = (const float*)d_in[base + 0];
    const float* bqkv = (const float*)d_in[base + 1];
    const float* wo   = (const float*)d_in[base + 2];
    const float* bo   = (const float*)d_in[base + 3];
    const float* g1   = (const float*)d_in[base + 4];
    const float* b1n  = (const float*)d_in[base + 5];
    const float* w1   = (const float*)d_in[base + 6];
    const float* b1   = (const float*)d_in[base + 7];
    const float* w2   = (const float*)d_in[base + 8];
    const float* b2   = (const float*)d_in[base + 9];
    const float* g2   = (const float*)d_in[base + 10];
    const float* b2n  = (const float*)d_in[base + 11];

    launch_gemm_nt(st, x, wqkv, bqkv, qkv, T, 3 * D_MODEL, D_MODEL, 0);
    dim3 ag(S / 16, N_HEAD, NC);
    attention_kernel<<<ag, 128, (size_t)96 * S, st>>>(qkv, mfix, ba, S);
    launch_gemm_nt(st, ba, wo, bo, bb, T, D_MODEL, D_MODEL, 0);
    ln_kernel<<<(T + 3) / 4, 128, 0, st>>>(x, bb, g1, b1n, x, T);
    launch_gemm_nt(st, x, w1, b1, qkv, T, FF_DIM, D_MODEL, 1);   // relu
    launch_gemm_nt(st, qkv, w2, b2, bb, T, D_MODEL, FF_DIM, 0);
    ln_kernel<<<(T + 3) / 4, 128, 0, st>>>(x, bb, g2, b2n, x, T);
  }
  mask_zero_kernel<<<T, D_MODEL, 0, st>>>(x, morig, T);
}

extern "C" void kernel_launch(void* const* d_in, const int* in_sizes, int n_in,
                              void* d_out, int out_size, void* d_ws, size_t ws_size,
                              hipStream_t stream) {
  (void)in_sizes; (void)n_in; (void)out_size; (void)ws_size;
  float* out = (float*)d_out;

  const int TD = BATCH * L_DRUG;    // 8192 drug tokens
  const int TP = BATCH * L_PROT;    // 65536 protein tokens

  // ---- workspace carve-up (float units) ----
  float* ws = (float*)d_ws;
  size_t o = 0;
  float* XD   = ws + o; o += (size_t)TD * D_MODEL;        // drug activations
  float* XP   = ws + o; o += (size_t)TP * D_MODEL;        // protein activations
  float* QKV  = ws + o; o += (size_t)TP * 3 * D_MODEL;    // qkv / ffn hidden / pool hidden
  float* BA   = ws + o; o += (size_t)TP * D_MODEL;        // attention output
  float* BB   = ws + o; o += (size_t)TP * D_MODEL;        // proj / ffn output
  float* SSC  = ws + o; o += (size_t)TP;                  // pooling scores
  float* REP  = ws + o; o += (size_t)BATCH * 256;         // concat [d_rep|p_rep]
  float* GH   = ws + o; o += (size_t)BATCH * D_MODEL;     // gate pre-LN
  float* GHN  = ws + o; o += (size_t)BATCH * D_MODEL;     // gate post-LN
  float* PROB = ws + o; o += 512;                         // probs [64,6]
  float* WMAT = ws + o; o += 512;                         // routing weights
  float* EH   = ws + o; o += (size_t)NEXP * BATCH * EXP_HID;
  float* EO   = ws + o; o += 512;                         // eo [6,64]
  int*   MFD  = (int*)(ws + o); o += (size_t)TD;          // fixed drug mask
  int*   MFP  = (int*)(ws + o); o += (size_t)TP;          // fixed prot mask

  const int*   dtok  = (const int*)d_in[IX_DTOK];
  const int*   ptok  = (const int*)d_in[IX_PTOK];
  const int*   dmask = (const int*)d_in[IX_DMASK];
  const int*   pmask = (const int*)d_in[IX_PMASK];
  const float* feats = (const float*)d_in[IX_FEATS];

  // ---- fix_all_pad masks (drug: 64 rows of 128; prot: 128 chunks of 512) ----
  fix_mask_kernel<<<1, 64, 0, stream>>>(dmask, MFD, BATCH, L_DRUG);
  fix_mask_kernel<<<1, 128, 0, stream>>>(pmask, MFP, BATCH * (L_PROT / CHUNK), CHUNK);

  // ---- drug encoder ----
  embed_kernel<<<TD, D_MODEL, 0, stream>>>(dtok, (const float*)d_in[IX_DEMB], XD, L_DRUG);
  { int lb[2] = {IX_DL0, IX_DL1};
    run_encoder(stream, XD, TD, L_DRUG, BATCH, MFD, dmask, QKV, BA, BB, d_in, lb); }

  // ---- protein encoder (128 independent 512-chunks) ----
  embed_kernel<<<TP, D_MODEL, 0, stream>>>(ptok, (const float*)d_in[IX_PEMB], XP, L_PROT);
  { int lb[2] = {IX_PL0, IX_PL1};
    run_encoder(stream, XP, TP, CHUNK, BATCH * (L_PROT / CHUNK), MFP, pmask,
                QKV, BA, BB, d_in, lb); }

  // ---- attention pooling -> REP = [d_rep | p_rep] ----
  launch_gemm_nt(stream, XD, (const float*)d_in[IX_DPOOL + 0],
                 (const float*)d_in[IX_DPOOL + 1], QKV, TD, 64, D_MODEL, 0);
  pool_score_kernel<<<TD / 256, 256, 0, stream>>>(QKV, (const float*)d_in[IX_DPOOL + 2],
                                                  (const float*)d_in[IX_DPOOL + 3], SSC, TD);
  pool_kernel<<<BATCH, 128, 0, stream>>>(XD, SSC, dmask, REP, L_DRUG, 0);

  launch_gemm_nt(stream, XP, (const float*)d_in[IX_PPOOL + 0],
                 (const float*)d_in[IX_PPOOL + 1], QKV, TP, 64, D_MODEL, 0);
  pool_score_kernel<<<TP / 256, 256, 0, stream>>>(QKV, (const float*)d_in[IX_PPOOL + 2],
                                                  (const float*)d_in[IX_PPOOL + 3], SSC, TP);
  pool_kernel<<<BATCH, 128, 0, stream>>>(XP, SSC, pmask, REP, L_PROT, D_MODEL);

  // ---- gating MLP + softmax + top-2 ----
  launch_gemm_nt(stream, REP, (const float*)d_in[IX_GATE + 0],
                 (const float*)d_in[IX_GATE + 1], GH, BATCH, D_MODEL, 2 * D_MODEL, 0);
  ln_kernel<<<(BATCH + 3) / 4, 128, 0, stream>>>(GH, nullptr,
                                                 (const float*)d_in[IX_GATE + 2],
                                                 (const float*)d_in[IX_GATE + 3], GHN, BATCH);
  gate_kernel<<<1, 64, 0, stream>>>(GHN, (const float*)d_in[IX_GATE + 4],
                                    (const float*)d_in[IX_GATE + 5], PROB, WMAT);

  // ---- experts: eh = relu(feats @ w1[e] + b1[e]) ; eo = eh . w2[e] + b2[e] ----
  const float* ew1 = (const float*)d_in[IX_EXP + 0];   // [6, 512, 1024]
  const float* eb1 = (const float*)d_in[IX_EXP + 1];   // [6, 1024]
  const float* ew2 = (const float*)d_in[IX_EXP + 2];   // [6, 1024]
  const float* eb2 = (const float*)d_in[IX_EXP + 3];   // [6]
  for (int e = 0; e < NEXP; ++e) {
    launch_gemm_nn(stream, feats, ew1 + (size_t)e * FEAT * EXP_HID,
                   eb1 + (size_t)e * EXP_HID, EH + (size_t)e * BATCH * EXP_HID,
                   BATCH, EXP_HID, FEAT, 1);
  }
  expert_reduce_kernel<<<dim3(BATCH, NEXP), 32, 0, stream>>>(EH, ew2, eb2, EO);

  // ---- aggregation + aux loss -> d_out[0:64]=final, d_out[64]=aux ----
  final_kernel<<<1, 64, 0, stream>>>(EO, WMAT,
                                     (const float*)d_in[IX_AGG + 0],
                                     (const float*)d_in[IX_AGG + 1],
                                     (const float*)d_in[IX_AGG + 2],
                                     (const float*)d_in[IX_AGG + 3], out);
  aux_kernel<<<1, 1, 0, stream>>>(PROB, out);
}
